// CharRNN_1305670058183
// MI455X (gfx1250) — compile-verified, matrix operand order checked by reading, not run on hardware
//
#include <hip/hip_runtime.h>
#include <hip/hip_bf16.h>

// ---------------------------------------------------------------------------
// CharRNN (2-layer BN-LSTM + softmax head) for MI455X / gfx1250, wave32.
// All matmuls via v_wmma_f32_16x16x32_bf16. Each wave owns a 16x64 output
// strip (4 accumulators). K-loop is manually ping-pong pipelined (two full
// fragment sets) so loads for the next 32-K step are in flight while the
// current step's 4 WMMAs execute.
// ---------------------------------------------------------------------------

typedef __attribute__((ext_vector_type(16))) __bf16 v16bf;
typedef __attribute__((ext_vector_type(8)))  __bf16 v8bf;
typedef __attribute__((ext_vector_type(8)))  float  v8f;

#define VOCAB 8000
#define NUNIT 256
#define HDIM  1024
#define BSZ   64
#define TLEN  128
#define G4    4096   // 4*HDIM

__device__ __forceinline__ float sigm(float x) { return 1.0f / (1.0f + __expf(-x)); }

// Load one 16x32 bf16 fragment (two 16B chunks per lane) per ISA 7.12.2 layout.
__device__ __forceinline__ v16bf ldfrag(const __bf16* __restrict__ p)
{
    v8bf lo = *(const v8bf*)(p);        // 16B aligned
    v8bf hi = *(const v8bf*)(p + 16);
    v16bf f;
#pragma unroll
    for (int e = 0; e < 8; ++e) { f[e] = lo[e]; f[8 + e] = hi[e]; }
    return f;
}

// ---------------------------------------------------------------------------
// WMMA GEMM: C[M,N] = A[M,K] * Bt[N,K]^T (+ bias[n]).
// A bf16 row-major (lda), Bt bf16 row-major [N][K] (ldb).
// Output: f32 to C (ldc) if Cb==nullptr, else bf16 to Cb (ldc).
// One wave -> one 16x64 strip (4 x 16x16 tiles); 8 waves per block.
// Requires M % 16 == 0, N % 64 == 0, K % 64 == 0.
// ---------------------------------------------------------------------------
__global__ void __launch_bounds__(256)
wmma_gemm_bf16(const __bf16* __restrict__ A, int lda,
               const __bf16* __restrict__ Bt, int ldb,
               float* __restrict__ C, __bf16* __restrict__ Cb, int ldc,
               const float* __restrict__ bias,
               int M, int N, int K)
{
    const int lane  = threadIdx.x & 31;
    const int wave  = threadIdx.x >> 5;
    const int NS    = N >> 6;                 // strips of 64 columns
    const int MT    = M >> 4;
    const int strip = blockIdx.x * (blockDim.x >> 5) + wave;
    if (strip >= MT * NS) return;             // wave-uniform: EXEC stays all-ones

    const int m0   = (strip / NS) << 4;
    const int n0   = (strip % NS) << 6;
    const int half = lane >> 4;               // 0 or 1
    const int l    = lane & 15;

    const __bf16* arow = A  + (size_t)(m0 + l) * lda + (half << 3);
    const __bf16* brow = Bt + (size_t)(n0 + l) * ldb + (half << 3);
    const size_t  bstp = (size_t)16 * ldb;    // next 16-column tile of the strip

    v8f acc0 = {}, acc1 = {}, acc2 = {}, acc3 = {};

    // ---- ping-pong pipelined K loop (two full fragment sets) --------------
    v16bf a0, b00, b01, b02, b03;             // set 0
    v16bf a1, b10, b11, b12, b13;             // set 1

    a0  = ldfrag(arow);
    b00 = ldfrag(brow);
    b01 = ldfrag(brow + bstp);
    b02 = ldfrag(brow + 2 * bstp);
    b03 = ldfrag(brow + 3 * bstp);

    for (int k0 = 0; k0 < K; k0 += 64) {
        // stream far-ahead lines toward the caches (global_prefetch_b8)
        if (k0 + 128 < K) {
            __builtin_prefetch(arow + k0 + 128, 0, 3);
            __builtin_prefetch(brow + k0 + 128, 0, 3);
            __builtin_prefetch(brow + bstp + k0 + 128, 0, 3);
            __builtin_prefetch(brow + 2 * bstp + k0 + 128, 0, 3);
            __builtin_prefetch(brow + 3 * bstp + k0 + 128, 0, 3);
        }
        // issue loads for set 1 (k0+32) before computing on set 0
        const int k1 = k0 + 32;
        a1  = ldfrag(arow + k1);
        b10 = ldfrag(brow + k1);
        b11 = ldfrag(brow + bstp + k1);
        b12 = ldfrag(brow + 2 * bstp + k1);
        b13 = ldfrag(brow + 3 * bstp + k1);

        acc0 = __builtin_amdgcn_wmma_f32_16x16x32_bf16(false, a0, false, b00, (short)0, acc0, false, false);
        acc1 = __builtin_amdgcn_wmma_f32_16x16x32_bf16(false, a0, false, b01, (short)0, acc1, false, false);
        acc2 = __builtin_amdgcn_wmma_f32_16x16x32_bf16(false, a0, false, b02, (short)0, acc2, false, false);
        acc3 = __builtin_amdgcn_wmma_f32_16x16x32_bf16(false, a0, false, b03, (short)0, acc3, false, false);

        // issue loads for the next set 0 (k0+64) before computing on set 1
        if (k0 + 64 < K) {
            const int k2 = k0 + 64;
            a0  = ldfrag(arow + k2);
            b00 = ldfrag(brow + k2);
            b01 = ldfrag(brow + bstp + k2);
            b02 = ldfrag(brow + 2 * bstp + k2);
            b03 = ldfrag(brow + 3 * bstp + k2);
        }

        acc0 = __builtin_amdgcn_wmma_f32_16x16x32_bf16(false, a1, false, b10, (short)0, acc0, false, false);
        acc1 = __builtin_amdgcn_wmma_f32_16x16x32_bf16(false, a1, false, b11, (short)0, acc1, false, false);
        acc2 = __builtin_amdgcn_wmma_f32_16x16x32_bf16(false, a1, false, b12, (short)0, acc2, false, false);
        acc3 = __builtin_amdgcn_wmma_f32_16x16x32_bf16(false, a1, false, b13, (short)0, acc3, false, false);
    }

    // C layout (ISA 7.12.2): VGPR v, lanes 0-15 -> M=m0+v, lanes 16-31 -> M=m0+v+8
    const size_t rbase = (size_t)(m0 + (half << 3)) * ldc;
#pragma unroll
    for (int j = 0; j < 4; ++j) {
        const v8f& acc = (j == 0) ? acc0 : (j == 1) ? acc1 : (j == 2) ? acc2 : acc3;
        const int  n   = n0 + (j << 4) + l;
        const float bv = bias ? bias[n] : 0.0f;
        if (Cb) {
            __bf16* crow = Cb + rbase + n;
#pragma unroll
            for (int v = 0; v < 8; ++v) crow[(size_t)v * ldc] = (__bf16)(acc[v] + bv);
        } else {
            float* crow = C + rbase + n;
#pragma unroll
            for (int v = 0; v < 8; ++v) crow[(size_t)v * ldc] = acc[v] + bv;
        }
    }
}

// ---------------------------------------------------------------------------
// BN-LSTM cell: one thread per hidden unit. Per-column batch stats over B=64
// of both pre-activation matrices, gates, c_new, BN(c_new); writes h_new as
// bf16 (input of the next WMMA GEMM).
// ---------------------------------------------------------------------------
__global__ void __launch_bounds__(256)
bnlstm_cell(const float* __restrict__ gxp,   // [B, 4H]  x @ Wx
            const float* __restrict__ ghp,   // [B, 4H]  h @ Wh
            const float* __restrict__ bvec,  // [4H]
            const float* __restrict__ gamx,  // [4H]
            const float* __restrict__ gamh,  // [4H]
            const float* __restrict__ gcv,   // [H]
            const float* __restrict__ bcv,   // [H]
            float*        __restrict__ c,    // [B, H] in/out
            __bf16*       __restrict__ hout) // [B, H]
{
    const int t = blockIdx.x * blockDim.x + threadIdx.x;
    if (t >= HDIM) return;
    const float eps = 1e-5f, inv = 1.0f / (float)BSZ;

    float mx[4], sx[4], mh[4], sh[4], bb[4];
#pragma unroll
    for (int g = 0; g < 4; ++g) {
        const int col = g * HDIM + t;
        float s0 = 0.f, q0 = 0.f, s1 = 0.f, q1 = 0.f;
        for (int r = 0; r < BSZ; ++r) {
            float a = gxp[(size_t)r * G4 + col];
            float b = ghp[(size_t)r * G4 + col];
            s0 += a; q0 += a * a; s1 += b; q1 += b * b;
        }
        float m0 = s0 * inv, v0 = q0 * inv - m0 * m0;
        float m1 = s1 * inv, v1 = q1 * inv - m1 * m1;
        mx[g] = m0; sx[g] = gamx[col] * rsqrtf(v0 + eps);
        mh[g] = m1; sh[g] = gamh[col] * rsqrtf(v1 + eps);
        bb[g] = bvec[col];
    }

    float cs = 0.f, cq = 0.f;
    for (int r = 0; r < BSZ; ++r) {
        const size_t rb = (size_t)r * G4;
        float gi = (gxp[rb + t]            - mx[0]) * sx[0] + (ghp[rb + t]            - mh[0]) * sh[0] + bb[0];
        float gj = (gxp[rb + HDIM + t]     - mx[1]) * sx[1] + (ghp[rb + HDIM + t]     - mh[1]) * sh[1] + bb[1];
        float gf = (gxp[rb + 2 * HDIM + t] - mx[2]) * sx[2] + (ghp[rb + 2 * HDIM + t] - mh[2]) * sh[2] + bb[2];
        float cn = sigm(gf + 1.0f) * c[(size_t)r * HDIM + t] + sigm(gi) * tanhf(gj);
        c[(size_t)r * HDIM + t] = cn;
        cs += cn; cq += cn * cn;
    }
    const float mc = cs * inv, vc = cq * inv - mc * mc;
    const float sc = gcv[t] * rsqrtf(vc + eps), bc = bcv[t];

    for (int r = 0; r < BSZ; ++r) {
        const size_t rb = (size_t)r * G4;
        float go = (gxp[rb + 3 * HDIM + t] - mx[3]) * sx[3] + (ghp[rb + 3 * HDIM + t] - mh[3]) * sh[3] + bb[3];
        float hn = sigm(go) * tanhf((c[(size_t)r * HDIM + t] - mc) * sc + bc);
        hout[(size_t)r * HDIM + t] = (__bf16)hn;
    }
}

// fp32 [K][N] -> bf16 [N][K] (transpose + convert, one-time per launch)
__global__ void convert_transpose(const float* __restrict__ in, __bf16* __restrict__ out,
                                  int K, int N)
{
    size_t i = (size_t)blockIdx.x * blockDim.x + threadIdx.x;
    if (i >= (size_t)K * N) return;
    int n = (int)(i % N);
    int k = (int)(i / N);
    out[(size_t)n * K + k] = (__bf16)in[i];
}

// embedding gather -> bf16 [T][B][NU]
__global__ void embed_gather(const int* __restrict__ ids, const float* __restrict__ emb,
                             __bf16* __restrict__ xb)
{
    size_t i = (size_t)blockIdx.x * blockDim.x + threadIdx.x;
    if (i >= (size_t)TLEN * BSZ * NUNIT) return;
    int u = (int)(i % NUNIT);
    size_t r = i / NUNIT;
    int b = (int)(r % BSZ);
    int t = (int)(r / BSZ);
    int id = ids[b * TLEN + t];
    xb[i] = (__bf16)emb[(size_t)id * NUNIT + u];
}

// ---------------------------------------------------------------------------
extern "C" void kernel_launch(void* const* d_in, const int* in_sizes, int n_in,
                              void* d_out, int out_size, void* d_ws, size_t ws_size,
                              hipStream_t stream)
{
    (void)in_sizes; (void)n_in; (void)out_size; (void)ws_size;

    const int*   ids  = (const int*)  d_in[0];
    const float* emb  = (const float*)d_in[1];
    const float* Wx0  = (const float*)d_in[2];
    const float* Wh0  = (const float*)d_in[3];
    const float* b0   = (const float*)d_in[4];
    const float* gx0  = (const float*)d_in[5];
    const float* gh0  = (const float*)d_in[6];
    const float* gc0  = (const float*)d_in[7];
    const float* bc0  = (const float*)d_in[8];
    const float* Wx1  = (const float*)d_in[9];
    const float* Wh1  = (const float*)d_in[10];
    const float* b1   = (const float*)d_in[11];
    const float* gx1  = (const float*)d_in[12];
    const float* gh1  = (const float*)d_in[13];
    const float* gc1  = (const float*)d_in[14];
    const float* bc1  = (const float*)d_in[15];
    const float* Wp   = (const float*)d_in[16];
    const float* bp   = (const float*)d_in[17];
    const float* smw  = (const float*)d_in[18];
    const float* smb  = (const float*)d_in[19];
    float* logits     = (float*)d_out;

    // workspace carve-up (all regions 256B aligned)
    char* ws = (char*)d_ws;
    size_t off = 0;
    auto carve = [&](size_t bytes) -> char* {
        char* p = ws + off;
        off += (bytes + 255) & ~(size_t)255;
        return p;
    };
    __bf16* xb    = (__bf16*)carve((size_t)TLEN * BSZ * NUNIT * 2);  // [T][B][NU]
    __bf16* Wx0t  = (__bf16*)carve((size_t)G4 * NUNIT * 2);          // [4H][NU]
    __bf16* Wh0t  = (__bf16*)carve((size_t)G4 * HDIM  * 2);          // [4H][H]
    __bf16* Wx1t  = (__bf16*)carve((size_t)G4 * HDIM  * 2);
    __bf16* Wh1t  = (__bf16*)carve((size_t)G4 * HDIM  * 2);
    __bf16* Wpt   = (__bf16*)carve((size_t)NUNIT * HDIM * 2);        // [NU][H]
    __bf16* SWt   = (__bf16*)carve((size_t)VOCAB * NUNIT * 2);       // [V][NU]
    float*  gxp   = (float*) carve((size_t)BSZ * G4 * 4);
    float*  ghp   = (float*) carve((size_t)BSZ * G4 * 4);
    __bf16* h0b   = (__bf16*)carve((size_t)BSZ * HDIM * 2);
    __bf16* h1b   = (__bf16*)carve((size_t)BSZ * HDIM * 2);
    float*  c0    = (float*) carve((size_t)BSZ * HDIM * 4);
    float*  c1    = (float*) carve((size_t)BSZ * HDIM * 4);
    __bf16* projb = (__bf16*)carve((size_t)BSZ * TLEN * NUNIT * 2);  // [B*T][NU] bf16

    const int TPB = 256;
    auto grid1d = [](size_t n, int tpb) { return (unsigned)((n + tpb - 1) / tpb); };

    // ---- one-time prep: bf16 weight transposes + embedding gather ----------
    convert_transpose<<<grid1d((size_t)NUNIT * G4,   TPB), TPB, 0, stream>>>(Wx0, Wx0t, NUNIT, G4);
    convert_transpose<<<grid1d((size_t)HDIM  * G4,   TPB), TPB, 0, stream>>>(Wh0, Wh0t, HDIM,  G4);
    convert_transpose<<<grid1d((size_t)HDIM  * G4,   TPB), TPB, 0, stream>>>(Wx1, Wx1t, HDIM,  G4);
    convert_transpose<<<grid1d((size_t)HDIM  * G4,   TPB), TPB, 0, stream>>>(Wh1, Wh1t, HDIM,  G4);
    convert_transpose<<<grid1d((size_t)HDIM * NUNIT, TPB), TPB, 0, stream>>>(Wp,  Wpt,  HDIM, NUNIT);
    convert_transpose<<<grid1d((size_t)NUNIT * VOCAB, TPB), TPB, 0, stream>>>(smw, SWt, NUNIT, VOCAB);
    embed_gather<<<grid1d((size_t)TLEN * BSZ * NUNIT, TPB), TPB, 0, stream>>>(ids, emb, xb);

    // zero recurrent state (graph-capture-safe async memset)
    hipMemsetAsync(h0b, 0, (size_t)BSZ * HDIM * 2, stream);
    hipMemsetAsync(h1b, 0, (size_t)BSZ * HDIM * 2, stream);
    hipMemsetAsync(c0,  0, (size_t)BSZ * HDIM * 4, stream);
    hipMemsetAsync(c1,  0, (size_t)BSZ * HDIM * 4, stream);

    // ---- sequential scan over T ------------------------------------------
    // strips = (M/16) * (N/64); 8 waves (strips) per block; all counts exact
    const int gBig  = (BSZ / 16) * (G4 / 64) / 8;     // 4*64/8  = 32 blocks
    const int gProj = (BSZ / 16) * (NUNIT / 64) / 8;  // 4*4/8   = 2 blocks
    for (int t = 0; t < TLEN; ++t) {
        const __bf16* xt = xb + (size_t)t * BSZ * NUNIT;
        // layer 0
        wmma_gemm_bf16<<<gBig, TPB, 0, stream>>>(xt,  NUNIT, Wx0t, NUNIT,
                                                 gxp, nullptr, G4, nullptr, BSZ, G4, NUNIT);
        wmma_gemm_bf16<<<gBig, TPB, 0, stream>>>(h0b, HDIM,  Wh0t, HDIM,
                                                 ghp, nullptr, G4, nullptr, BSZ, G4, HDIM);
        bnlstm_cell<<<HDIM / TPB, TPB, 0, stream>>>(gxp, ghp, b0, gx0, gh0, gc0, bc0, c0, h0b);
        // layer 1
        wmma_gemm_bf16<<<gBig, TPB, 0, stream>>>(h0b, HDIM, Wx1t, HDIM,
                                                 gxp, nullptr, G4, nullptr, BSZ, G4, HDIM);
        wmma_gemm_bf16<<<gBig, TPB, 0, stream>>>(h1b, HDIM, Wh1t, HDIM,
                                                 ghp, nullptr, G4, nullptr, BSZ, G4, HDIM);
        bnlstm_cell<<<HDIM / TPB, TPB, 0, stream>>>(gxp, ghp, b1, gx1, gh1, gc1, bc1, c1, h1b);
        // projection -> bf16 rows at (b*T + t) of the [B*T, NU] activation matrix
        wmma_gemm_bf16<<<gProj, TPB, 0, stream>>>(h1b, HDIM, Wpt, HDIM,
                                                  nullptr, projb + (size_t)t * NUNIT,
                                                  TLEN * NUNIT, bp, BSZ, NUNIT, HDIM);
    }

    // ---- softmax head: [8192x256] @ [256x8000] + b -> d_out (f32) ---------
    const int gHead = ((BSZ * TLEN) / 16) * (VOCAB / 64) / 8;  // 512*125/8 = 8000
    wmma_gemm_bf16<<<gHead, TPB, 0, stream>>>(projb, NUNIT, SWt, NUNIT,
                                              logits, nullptr, VOCAB, smb,
                                              BSZ * TLEN, VOCAB, NUNIT);
}
